// ProteinGATv2_37958920962627
// MI455X (gfx1250) — matrix operand order repaired and checked by reading, not compiled, not run on hardware
//
#include <hip/hip_runtime.h>
#include <hip/hip_bf16.h>
#include <math.h>

// Problem constants (match reference)
#define Nn  50000
#define Ee  800000
#define INF_ 64
#define HID 128
#define Hh  4
#define Cc  32
#define Ll  3
#define EDd 16
#define Gg  32
#define OUTD 128

typedef __attribute__((ext_vector_type(16))) _Float16 v16h;
typedef __attribute__((ext_vector_type(8)))  float    v8f;

// ---------------------------------------------------------------------------
// helpers
// ---------------------------------------------------------------------------
__device__ __forceinline__ void atomicMaxFloat(float* addr, float val) {
    // sign-aware int/uint trick: valid for values initialized to -inf
    if (val >= 0.0f) atomicMax((int*)addr, __float_as_int(val));
    else             atomicMin((unsigned int*)addr, __float_as_uint(val));
}

__global__ void fill_f32(float* __restrict__ p, float v, long n) {
    long i = (long)blockIdx.x * blockDim.x + threadIdx.x;
    if (i < n) p[i] = v;
}

// ---------------------------------------------------------------------------
// WMMA GEMM: C[M,Nc] = A[M,K] @ B[K,Nc] (+bias), f32 in/out, f16 compute.
// One wave per 16-column strip; wave preloads ALL B (weight) fragments into
// registers once, then sweeps RT=4 consecutive 16-row M tiles reusing them.
// A fragments are loaded as float2 (b64) — layout pairs are contiguous and
// 8B aligned. K is a compile-time multiple of 32 so everything unrolls into
// straight v_wmma chains. Requires M%16==0, Nc%16==0.
// ---------------------------------------------------------------------------
template <int K>
__global__ __launch_bounds__(128) void wmma_gemm(
    const float* __restrict__ A, const float* __restrict__ B,
    const float* __restrict__ bias, float* __restrict__ C, int M, int Nc)
{
    constexpr int KS = K / 32;   // k-steps
    constexpr int RT = 4;        // row tiles per wave

    const int wid  = threadIdx.x >> 5;
    const int lane = threadIdx.x & 31;
    const int tn   = blockIdx.y * 4 + wid;
    if (tn * 16 >= Nc) return;           // whole-wave uniform exit (EXEC stays full)

    const int hs   = lane >> 4;          // half-select (lanes 0-15 vs 16-31)
    const int ncol = tn * 16 + (lane & 15);

    // ---- preload all B fragments (weights, L2-hot) once per wave ----
    v16h bf[KS];
#pragma unroll
    for (int ks = 0; ks < KS; ++ks)
#pragma unroll
        for (int j = 0; j < 16; ++j) {
            // B 32x16 f16 fragment: k = j + hs*16, n = lane&15
            const int kb = ks * 32 + j + (hs << 4);
            bf[ks][j] = (_Float16)B[(long)kb * Nc + ncol];
        }

    const float bv = bias ? bias[ncol] : 0.0f;

#pragma unroll
    for (int t = 0; t < RT; ++t) {
        const int tm = blockIdx.x * RT + t;
        if (tm * 16 < M) {               // wave-uniform guard
            const int mrow = tm * 16 + (lane & 15);
            v8f acc = {};
#pragma unroll
            for (int ks = 0; ks < KS; ++ks) {
                v16h a;
#pragma unroll
                for (int p = 0; p < 8; ++p) {
                    // pair (j=2p, 2p+1): ka = 2p + hs*16, contiguous
                    const int ka = ks * 32 + (p << 1) + (hs << 4);
                    const float2 av =
                        *(const float2*)(A + (long)mrow * K + ka);
                    a[2 * p]     = (_Float16)av.x;
                    a[2 * p + 1] = (_Float16)av.y;
                }
                acc = __builtin_amdgcn_wmma_f32_16x16x32_f16(
                    false, a, false, bf[ks], (short)0, acc, false, false);
            }
#pragma unroll
            for (int r = 0; r < 8; ++r) {
                const int row = tm * 16 + r + hs * 8;  // C/D: M = r (+8 hi lanes)
                C[(long)row * Nc + ncol] = acc[r] + bv;
            }
        }
    }
}

// ---------------------------------------------------------------------------
// Edge pass 1: per-edge GATv2 logits + segment-max over dst.
// One wave per edge; lane owns channel (h*32 + lane) for each of 4 heads.
// ee recomputed on the fly (16x128 matvec); xl/xr rows prefetched so the L2
// gather latency hides behind that VALU work.
// ---------------------------------------------------------------------------
__global__ __launch_bounds__(256) void edge_logits(
    const int* __restrict__ src, const int* __restrict__ dst,
    const float* __restrict__ ea,   // [E,16]
    const float* __restrict__ We,   // [16,128] (layer slice)
    const float* __restrict__ att,  // [4,32]   (layer slice)
    const float* __restrict__ xl, const float* __restrict__ xr,
    float* __restrict__ logits,     // [E,4]
    float* __restrict__ lmax,       // [N,4] (init -inf)
    int E_)
{
    const int wid = threadIdx.x >> 5, lane = threadIdx.x & 31;
    const long e = (long)blockIdx.x * 8 + wid;
    if (e >= E_) return;
    const int s = src[e], d = dst[e];

    // each lane prefetches a 16B slice -> full 512B row covered per wave
    __builtin_prefetch(xl + (long)s * HID + lane * 4, 0, 3);
    __builtin_prefetch(xr + (long)d * HID + lane * 4, 0, 3);

    float part[Hh];
#pragma unroll
    for (int h = 0; h < Hh; ++h) {
        const int ch = h * Cc + lane;
        float eev = 0.0f;
#pragma unroll
        for (int dd = 0; dd < EDd; ++dd)
            eev += ea[e * EDd + dd] * We[dd * HID + ch];
        float mv = xl[(long)s * HID + ch] + xr[(long)d * HID + ch] + eev;
        mv = mv > 0.0f ? mv : 0.2f * mv;          // leaky_relu(0.2)
        part[h] = mv * att[h * Cc + lane];
    }
#pragma unroll
    for (int off = 16; off > 0; off >>= 1)
#pragma unroll
        for (int h = 0; h < Hh; ++h)
            part[h] += __shfl_xor(part[h], off, 32);

    if (lane == 0) {
#pragma unroll
        for (int h = 0; h < Hh; ++h) {
            logits[e * Hh + h] = part[h];
            atomicMaxFloat(&lmax[(long)d * Hh + h], part[h]);
        }
    }
}

// Edge pass 2: ex = exp(logit - lmax[dst]); denom[dst] += ex. Thread per (e,h).
__global__ void edge_softmax_num(
    const int* __restrict__ dst, const float* __restrict__ lmax,
    float* __restrict__ logits, float* __restrict__ denom, long EH)
{
    const long i = (long)blockIdx.x * blockDim.x + threadIdx.x;
    if (i >= EH) return;
    const long e = i >> 2; const int h = (int)(i & 3);
    const int d = dst[e];
    const float ex = __expf(logits[i] - lmax[(long)d * Hh + h]);
    logits[i] = ex;
    atomicAdd(&denom[(long)d * Hh + h], ex);
}

// Edge pass 3: out[dst] += alpha * xl[src]. One wave per edge.
__global__ __launch_bounds__(256) void edge_aggregate(
    const int* __restrict__ src, const int* __restrict__ dst,
    const float* __restrict__ xl, const float* __restrict__ exbuf,
    const float* __restrict__ denom, float* __restrict__ outbuf, int E_)
{
    const int wid = threadIdx.x >> 5, lane = threadIdx.x & 31;
    const long e = (long)blockIdx.x * 8 + wid;
    if (e >= E_) return;
    const int s = src[e], d = dst[e];
    __builtin_prefetch(xl + (long)s * HID + lane * 4, 0, 3);
#pragma unroll
    for (int h = 0; h < Hh; ++h) {
        const float alpha = exbuf[e * Hh + h] /
                            (denom[(long)d * Hh + h] + 1e-16f);
        const int ch = h * Cc + lane;
        atomicAdd(&outbuf[(long)d * HID + ch], alpha * xl[(long)s * HID + ch]);
    }
}

// Post: h = relu(LN(agg + conv_b + h)); also write into jkcat column block.
__global__ __launch_bounds__(256) void post_ln(
    const float* __restrict__ agg, const float* __restrict__ conv_b,
    const float* __restrict__ ln_g, const float* __restrict__ ln_b,
    float* __restrict__ h, float* __restrict__ jkcat, int layer, int N_)
{
    const int wid = threadIdx.x >> 5, lane = threadIdx.x & 31;
    const long n = (long)blockIdx.x * 8 + wid;
    if (n >= N_) return;
    float v[4]; float sum = 0.0f;
#pragma unroll
    for (int q = 0; q < 4; ++q) {
        const int ch = q * 32 + lane;
        v[q] = agg[n * HID + ch] + conv_b[ch] + h[n * HID + ch];
        sum += v[q];
    }
    for (int off = 16; off > 0; off >>= 1) sum += __shfl_xor(sum, off, 32);
    const float mu = sum * (1.0f / HID);
    float vs = 0.0f;
#pragma unroll
    for (int q = 0; q < 4; ++q) { const float t = v[q] - mu; vs += t * t; }
    for (int off = 16; off > 0; off >>= 1) vs += __shfl_xor(vs, off, 32);
    const float rstd = rsqrtf(vs * (1.0f / HID) + 1e-5f);
#pragma unroll
    for (int q = 0; q < 4; ++q) {
        const int ch = q * 32 + lane;
        float y = (v[q] - mu) * rstd * ln_g[ch] + ln_b[ch];
        y = fmaxf(y, 0.0f);
        h[n * HID + ch] = y;
        jkcat[n * (Ll * HID) + layer * HID + ch] = y;
    }
}

// Readout pass 1: scores = node @ ro_att_w ; smax[g] = max. Wave per node.
__global__ __launch_bounds__(256) void readout_scores(
    const float* __restrict__ node, const float* __restrict__ ro_w,
    const int* __restrict__ batch, float* __restrict__ scores,
    float* __restrict__ smax, int N_)
{
    const int wid = threadIdx.x >> 5, lane = threadIdx.x & 31;
    const long n = (long)blockIdx.x * 8 + wid;
    if (n >= N_) return;
    float s = 0.0f;
#pragma unroll
    for (int q = 0; q < 4; ++q) {
        const int ch = q * 32 + lane;
        s += node[n * OUTD + ch] * ro_w[ch];
    }
    for (int off = 16; off > 0; off >>= 1) s += __shfl_xor(s, off, 32);
    if (lane == 0) {
        scores[n] = s;
        atomicMaxFloat(&smax[batch[n]], s);
    }
}

// Readout pass 2: ex + denom. Thread per node.
__global__ void readout_ex(
    const int* __restrict__ batch, const float* __restrict__ smax,
    float* __restrict__ scores, float* __restrict__ denomg, int N_)
{
    const long n = (long)blockIdx.x * blockDim.x + threadIdx.x;
    if (n >= N_) return;
    const int g = batch[n];
    const float ex = __expf(scores[n] - smax[g]);
    scores[n] = ex;
    atomicAdd(&denomg[g], ex);
}

// Readout pass 3: pooled[g] += w[n]*node[n]. Thread = channel; block walks a
// chunk of batch-sorted nodes, accumulating in registers, flushing per graph.
__global__ __launch_bounds__(128) void pool_accum(
    const float* __restrict__ node, const int* __restrict__ batch,
    const float* __restrict__ ex, const float* __restrict__ denomg,
    float* __restrict__ pooled, int N_, int chunk)
{
    const int ch = threadIdx.x;
    long start = (long)blockIdx.x * chunk;
    long end = start + chunk; if (end > N_) end = N_;
    float acc = 0.0f; int cur = -1;
    for (long n = start; n < end; ++n) {
        if (n + 8 < end) __builtin_prefetch(node + (n + 8) * OUTD + ch, 0, 3);
        const int g = batch[n];
        if (g != cur) {
            if (cur >= 0) atomicAdd(&pooled[(long)cur * OUTD + ch], acc);
            acc = 0.0f; cur = g;
        }
        const float w = ex[n] / (denomg[g] + 1e-16f);
        acc += w * node[n * OUTD + ch];
    }
    if (cur >= 0) atomicAdd(&pooled[(long)cur * OUTD + ch], acc);
}

// ---------------------------------------------------------------------------
extern "C" void kernel_launch(void* const* d_in, const int* in_sizes, int n_in,
                              void* d_out, int out_size, void* d_ws, size_t ws_size,
                              hipStream_t stream)
{
    const float* x      = (const float*)d_in[0];
    const int*   ei     = (const int*)  d_in[1];
    const float* ea     = (const float*)d_in[2];
    const int*   batch  = (const int*)  d_in[3];
    const float* in_W   = (const float*)d_in[4];
    const float* in_b   = (const float*)d_in[5];
    const float* W_src  = (const float*)d_in[6];
    const float* W_dst  = (const float*)d_in[7];
    const float* W_edge = (const float*)d_in[8];
    const float* att    = (const float*)d_in[9];
    const float* conv_b = (const float*)d_in[10];
    const float* ln_g   = (const float*)d_in[11];
    const float* ln_b   = (const float*)d_in[12];
    const float* jk_W   = (const float*)d_in[13];
    const float* jk_b   = (const float*)d_in[14];
    const float* ro_att = (const float*)d_in[15];
    const float* ro_W   = (const float*)d_in[16];
    const float* ro_b   = (const float*)d_in[17];

    const int* src = ei;          // edge_index[0]
    const int* dst = ei + Ee;     // edge_index[1]

    float* nodeOut  = (float*)d_out;                     // [N,128]
    float* graphOut = (float*)d_out + (size_t)Nn * OUTD; // [32,128]

    // workspace layout (floats)
    float* ws = (float*)d_ws;
    size_t o = 0;
    float* h      = ws + o; o += (size_t)Nn * HID;
    float* xl     = ws + o; o += (size_t)Nn * HID;
    float* xr     = ws + o; o += (size_t)Nn * HID;  // reused as agg after logits
    float* logits = ws + o; o += (size_t)Ee * Hh;
    float* lmax   = ws + o; o += (size_t)Nn * Hh;
    float* denom  = ws + o; o += (size_t)Nn * Hh;
    float* jkcat  = ws + o; o += (size_t)Nn * Ll * HID;
    float* scores = ws + o; o += (size_t)Nn;
    float* smax   = ws + o; o += (size_t)Gg;
    float* denomg = ws + o; o += (size_t)Gg;
    float* pooled = ws + o; o += (size_t)Gg * OUTD;

    auto fill = [&](float* p, float v, long n) {
        fill_f32<<<(unsigned)((n + 255) / 256), 256, 0, stream>>>(p, v, n);
    };

    const dim3 gemmBlk(128);
    // grid.x covers M in groups of 4 tiles (64 rows) per wave
    const dim3 gNodeH((Nn / 16 + 3) / 4, (HID / 16 + 3) / 4);
    const unsigned egrid = (unsigned)((Ee + 7) / 8);
    const unsigned ngrid = (unsigned)((Nn + 7) / 8);

    // 1) input projection: h = x @ in_W + in_b   (K=64)
    wmma_gemm<INF_><<<gNodeH, gemmBlk, 0, stream>>>(x, in_W, in_b, h, Nn, HID);

    // 2) GATv2 layers
    for (int l = 0; l < Ll; ++l) {
        const float* Ws = W_src + (size_t)l * HID * HID;
        const float* Wd = W_dst + (size_t)l * HID * HID;
        const float* We = W_edge + (size_t)l * EDd * HID;
        const float* al = att + (size_t)l * Hh * Cc;

        wmma_gemm<HID><<<gNodeH, gemmBlk, 0, stream>>>(h, Ws, nullptr, xl, Nn, HID);
        wmma_gemm<HID><<<gNodeH, gemmBlk, 0, stream>>>(h, Wd, nullptr, xr, Nn, HID);

        fill(lmax, -INFINITY, (long)Nn * Hh);
        fill(denom, 0.0f, (long)Nn * Hh);

        edge_logits<<<egrid, 256, 0, stream>>>(src, dst, ea, We, al, xl, xr,
                                               logits, lmax, Ee);

        float* agg = xr;                       // xr dead after logits
        fill(agg, 0.0f, (long)Nn * HID);

        const long EH = (long)Ee * Hh;
        edge_softmax_num<<<(unsigned)((EH + 255) / 256), 256, 0, stream>>>(
            dst, lmax, logits, denom, EH);

        edge_aggregate<<<egrid, 256, 0, stream>>>(src, dst, xl, logits, denom,
                                                  agg, Ee);

        post_ln<<<ngrid, 256, 0, stream>>>(agg, conv_b + (size_t)l * HID,
                                           ln_g + (size_t)l * HID,
                                           ln_b + (size_t)l * HID,
                                           h, jkcat, l, Nn);
    }

    // 3) JK 'cat' projection: node = jkcat @ jk_W + jk_b   (K=384)
    wmma_gemm<Ll * HID><<<gNodeH, gemmBlk, 0, stream>>>(jkcat, jk_W, jk_b,
                                                        nodeOut, Nn, OUTD);

    // 4) attention-pool readout
    fill(smax, -INFINITY, Gg);
    fill(denomg, 0.0f, Gg);
    fill(pooled, 0.0f, (long)Gg * OUTD);

    readout_scores<<<ngrid, 256, 0, stream>>>(nodeOut, ro_att, batch, scores,
                                              smax, Nn);
    readout_ex<<<(unsigned)((Nn + 255) / 256), 256, 0, stream>>>(batch, smax,
                                                                 scores, denomg, Nn);
    const int chunk = 512;
    pool_accum<<<(unsigned)((Nn + chunk - 1) / chunk), 128, 0, stream>>>(
        nodeOut, batch, scores, denomg, pooled, Nn, chunk);

    // 5) graph = pooled @ ro_W + ro_b   (M=32, K=128)
    dim3 gG((Gg / 16 + 3) / 4, (OUTD / 16 + 3) / 4);
    wmma_gemm<HID><<<gG, gemmBlk, 0, stream>>>(pooled, ro_W, ro_b, graphOut,
                                               Gg, OUTD);
}
